// DurationBiasedAttention_21672404976047
// MI455X (gfx1250) — matrix operand order repaired and checked by reading, not compiled, not run on hardware
//
#include <hip/hip_runtime.h>
#include <hip/hip_bf16.h>

typedef __attribute__((ext_vector_type(16))) _Float16 v16h;
typedef __attribute__((ext_vector_type(8)))  float    v8f;

#define WMMA_F16(A, B, C) \
  __builtin_amdgcn_wmma_f32_16x16x32_f16(false, (A), false, (B), (short)0, (C), false, false)

// Problem constants
#define BATCH 2
#define LSEQ  2048
#define HIDN  512
#define NHEAD 8
#define HDIM  64
#define NTOK  (BATCH * LSEQ)   // 4096
#define NEGI  (-1.0e9f)

// ---------------------------------------------------------------------------
// Convert + transpose 4 weight matrices (512x512 f32, in-major) to f16 out-major
// dst layout: [w][o][i], w in {q,k,v,o}
// ---------------------------------------------------------------------------
__global__ void k_cvt_w(const float* __restrict__ Wq, const float* __restrict__ Wk,
                        const float* __restrict__ Wv, const float* __restrict__ Wo,
                        _Float16* __restrict__ dst) {
  int idx = blockIdx.x * 256 + threadIdx.x;     // 4*512*512 elements
  int w = idx >> 18;
  int r = idx & 262143;
  int o = r >> 9;
  int i = r & 511;
  const float* W = (w == 0) ? Wq : (w == 1) ? Wk : (w == 2) ? Wv : Wo;
  dst[idx] = (_Float16)W[i * 512 + o];
}

__global__ void k_cvt_x(const float* __restrict__ x, _Float16* __restrict__ xh) {
  int idx = blockIdx.x * 256 + threadIdx.x;     // NTOK*HIDN
  xh[idx] = (_Float16)x[idx];
}

// ---------------------------------------------------------------------------
// QKV projection: 8 waves per block share one Wt output-panel (stays in L0).
// Each wave: one 16x16 output tile, 16 WMMAs over K=512.
// Q,K stored [bh][l][d] (f16); V stored transposed [bh][d][l] (f16).
// ---------------------------------------------------------------------------
__global__ void k_proj(const _Float16* __restrict__ xh, const _Float16* __restrict__ WT,
                       const float* __restrict__ bq, const float* __restrict__ bk,
                       const float* __restrict__ bv,
                       _Float16* __restrict__ Qd, _Float16* __restrict__ Kd,
                       _Float16* __restrict__ Vt) {
  const int lane  = threadIdx.x & 31;
  const int warp  = threadIdx.x >> 5;
  const int row16 = lane & 15;
  const int hi    = lane >> 4;
  const int tt = blockIdx.x * 8 + warp;   // token tile (16 tokens)
  const int ot = blockIdx.y;              // output-column tile (16 cols)
  const int z  = blockIdx.z;              // 0=Q 1=K 2=V

  const _Float16* Wz = WT + (size_t)z * (512 * 512);
  const float* bz = (z == 0) ? bq : (z == 1) ? bk : bv;

  const int tok  = tt * 16 + row16;
  const int ocol = ot * 16 + row16;

  const _Float16* arow = xh + (size_t)tok * 512 + hi * 16;   // A operand rows
  const _Float16* brow = Wz + (size_t)ocol * 512 + hi * 16;  // B operand cols (contiguous)

  v8f acc = {};
  for (int kk = 0; kk < 512; kk += 32) {
    v16h a, b;
#pragma unroll
    for (int i = 0; i < 16; ++i) { a[i] = arow[kk + i]; }
#pragma unroll
    for (int i = 0; i < 16; ++i) { b[i] = brow[kk + i]; }
    acc = WMMA_F16(a, b, acc);
  }

  const float bb = bz[ocol];
  const int hh = ocol >> 6;        // head of this out column
  const int dd = ocol & 63;        // dim within head
#pragma unroll
  for (int r = 0; r < 8; ++r) {
    int t = tt * 16 + r + 8 * hi;          // global token
    int bidx = t >> 11;                    // batch
    int l = t & 2047;                      // position
    int bh = bidx * NHEAD + hh;
    _Float16 val = (_Float16)(acc[r] + bb);
    if (z == 0)      Qd[((size_t)bh * LSEQ + l) * HDIM + dd] = val;
    else if (z == 1) Kd[((size_t)bh * LSEQ + l) * HDIM + dd] = val;
    else             Vt[((size_t)bh * HDIM + dd) * LSEQ + l] = val;   // transposed
  }
}

// ---------------------------------------------------------------------------
// Flash-style biased attention.
// Block = 8 waves = 128 queries of ONE (b,h); all waves walk the same 32-key
// tiles in lockstep (the two per-iteration barriers enforce it), so K/V loads
// hit the WGP/L0 cache instead of re-streaming from L2 per wave.
// Online softmax, biases + mask fused, P@V accumulated in 4 x v8f (16x64 f32).
// ---------------------------------------------------------------------------
__global__ void k_attn(const _Float16* __restrict__ Qd, const _Float16* __restrict__ Kd,
                       const _Float16* __restrict__ Vt,
                       const int* __restrict__ amask, const int* __restrict__ gids,
                       const int* __restrict__ dids,
                       const float* __restrict__ gap_emb, const float* __restrict__ dur_emb,
                       _Float16* __restrict__ ctx) {
  const int lane  = threadIdx.x & 31;
  const int warp  = threadIdx.x >> 5;
  const int row16 = lane & 15;
  const int hi    = lane >> 4;
  const int bh = blockIdx.x;            // 0..15
  const int b = bh >> 3;
  const int h = bh & 7;
  const int q0 = (blockIdx.y * 8 + warp) * 16;

  const _Float16* Qb = Qd + (size_t)bh * LSEQ * HDIM;
  const _Float16* Kb = Kd + (size_t)bh * LSEQ * HDIM;
  const _Float16* Vb = Vt + (size_t)bh * HDIM * LSEQ;

  // Load Q tile (16x64) in A-operand layout: two d-chunks of 32
  v16h aq[2];
#pragma unroll
  for (int c = 0; c < 2; ++c) {
    const _Float16* p = Qb + (size_t)(q0 + row16) * HDIM + c * 32 + hi * 16;
#pragma unroll
    for (int i = 0; i < 16; ++i) { aq[c][i] = p[i]; }
  }

  // Per-lane copy of the 8 query time-gap ids this half-wave owns
  int gq[8];
#pragma unroll
  for (int r = 0; r < 8; ++r) gq[r] = gids[b * LSEQ + q0 + r + 8 * hi];

  float m_run[8], l_run[8];
#pragma unroll
  for (int r = 0; r < 8; ++r) { m_run[r] = -3.0e38f; l_run[r] = 0.0f; }

  v8f acc[4];
#pragma unroll
  for (int n = 0; n < 4; ++n) acc[n] = (v8f){};

  __shared__ _Float16 Pl[8 * 16 * 32];   // per-warp P staging (C-layout -> A-layout)
  _Float16* Pw = &Pl[warp * (16 * 32)];

  for (int k0 = 0; k0 < LSEQ; k0 += 32) {
    // ---- S = Q K^T for 32 keys (two 16-key WMMA tiles, two d-chunks each)
    v8f s[2];
#pragma unroll
    for (int j = 0; j < 2; ++j) {
      const _Float16* kp = Kb + (size_t)(k0 + j * 16 + row16) * HDIM + hi * 16;
      v16h bk0, bk1;
#pragma unroll
      for (int i = 0; i < 16; ++i) { bk0[i] = kp[i]; bk1[i] = kp[32 + i]; }
      v8f cz = (v8f){};
      cz = WMMA_F16(aq[0], bk0, cz);
      cz = WMMA_F16(aq[1], bk1, cz);
      s[j] = cz;
    }

    // ---- biases + mask
    float sc[2][8];
#pragma unroll
    for (int j = 0; j < 2; ++j) {
      int key = k0 + j * 16 + row16;
      int gk = gids[b * LSEQ + key];
      float db = dur_emb[dids[b * LSEQ + key] * NHEAD + h];
      int mk = amask[b * LSEQ + key];
#pragma unroll
      for (int r = 0; r < 8; ++r) {
        int rel = gq[r] - gk; rel = (rel < 0) ? -rel : rel; if (rel > 32) rel = 32;
        float v = s[j][r] * 0.125f + gap_emb[rel * NHEAD + h] + db;
        sc[j][r] = mk ? v : NEGI;
      }
    }

    // ---- online softmax (row = across 16 lanes within each half-wave)
    float tmax[8];
#pragma unroll
    for (int r = 0; r < 8; ++r) tmax[r] = fmaxf(sc[0][r], sc[1][r]);
#pragma unroll
    for (int off = 1; off < 16; off <<= 1)
#pragma unroll
      for (int r = 0; r < 8; ++r) tmax[r] = fmaxf(tmax[r], __shfl_xor(tmax[r], off, 32));

    float newm[8], alpha[8], p[2][8], tsum[8];
#pragma unroll
    for (int r = 0; r < 8; ++r) {
      newm[r] = fmaxf(m_run[r], tmax[r]);
      alpha[r] = __expf(m_run[r] - newm[r]);
      m_run[r] = newm[r];
      p[0][r] = __expf(sc[0][r] - newm[r]);
      p[1][r] = __expf(sc[1][r] - newm[r]);
      tsum[r] = p[0][r] + p[1][r];
    }
#pragma unroll
    for (int off = 1; off < 16; off <<= 1)
#pragma unroll
      for (int r = 0; r < 8; ++r) tsum[r] += __shfl_xor(tsum[r], off, 32);
#pragma unroll
    for (int r = 0; r < 8; ++r) l_run[r] = l_run[r] * alpha[r] + tsum[r];
#pragma unroll
    for (int n = 0; n < 4; ++n)
#pragma unroll
      for (int r = 0; r < 8; ++r) acc[n][r] *= alpha[r];

    // ---- stage P (16x32) in LDS: C-layout writes, A-layout reads
    __syncthreads();   // also keeps the 8 waves in lockstep for K/V cache reuse
#pragma unroll
    for (int j = 0; j < 2; ++j)
#pragma unroll
      for (int r = 0; r < 8; ++r)
        Pw[(r + 8 * hi) * 32 + j * 16 + row16] = (_Float16)p[j][r];
    __syncthreads();

    v16h ap;
    {
      const _Float16* pp = &Pw[row16 * 32 + hi * 16];
#pragma unroll
      for (int i = 0; i < 16; ++i) { ap[i] = pp[i]; }
    }

    // ---- O += P @ V  (V is stored transposed: contiguous B operand)
#pragma unroll
    for (int n = 0; n < 4; ++n) {
      const _Float16* vp = Vb + (size_t)(n * 16 + row16) * LSEQ + k0 + hi * 16;
      v16h bvv;
#pragma unroll
      for (int i = 0; i < 16; ++i) bvv[i] = vp[i];
      acc[n] = WMMA_F16(ap, bvv, acc[n]);
    }
  }

  // ---- finalize: divide by l_run, write context (f16, token-major)
#pragma unroll
  for (int n = 0; n < 4; ++n)
#pragma unroll
    for (int r = 0; r < 8; ++r) {
      float o = acc[n][r] / l_run[r];
      size_t tokrow = (size_t)(b * LSEQ + q0 + r + 8 * hi);
      int col = h * HDIM + n * 16 + row16;
      ctx[tokrow * HIDN + col] = (_Float16)o;
    }
}

// ---------------------------------------------------------------------------
// Output projection: ctx(f16) @ Wo + bo -> f32 out. 8 waves share one B panel.
// ---------------------------------------------------------------------------
__global__ void k_oproj(const _Float16* __restrict__ ctx, const _Float16* __restrict__ WoT,
                        const float* __restrict__ bo, float* __restrict__ out) {
  const int lane  = threadIdx.x & 31;
  const int warp  = threadIdx.x >> 5;
  const int row16 = lane & 15;
  const int hi    = lane >> 4;
  const int tt = blockIdx.x * 8 + warp;
  const int ot = blockIdx.y;
  const int tok  = tt * 16 + row16;
  const int ocol = ot * 16 + row16;

  const _Float16* arow = ctx + (size_t)tok * 512 + hi * 16;
  const _Float16* brow = WoT + (size_t)ocol * 512 + hi * 16;

  v8f acc = {};
  for (int kk = 0; kk < 512; kk += 32) {
    v16h a, b;
#pragma unroll
    for (int i = 0; i < 16; ++i) { a[i] = arow[kk + i]; }
#pragma unroll
    for (int i = 0; i < 16; ++i) { b[i] = brow[kk + i]; }
    acc = WMMA_F16(a, b, acc);
  }
  const float bb = bo[ocol];
#pragma unroll
  for (int r = 0; r < 8; ++r)
    out[(size_t)(tt * 16 + r + 8 * hi) * 512 + ocol] = acc[r] + bb;
}

// ---------------------------------------------------------------------------
extern "C" void kernel_launch(void* const* d_in, const int* in_sizes, int n_in,
                              void* d_out, int out_size, void* d_ws, size_t ws_size,
                              hipStream_t stream) {
  const float* x        = (const float*)d_in[0];
  const int*   amask    = (const int*)d_in[1];
  const int*   gids     = (const int*)d_in[2];
  const int*   dids     = (const int*)d_in[3];
  const float* Wq       = (const float*)d_in[4];
  const float* bq       = (const float*)d_in[5];
  const float* Wk       = (const float*)d_in[6];
  const float* bk       = (const float*)d_in[7];
  const float* Wv       = (const float*)d_in[8];
  const float* bv       = (const float*)d_in[9];
  const float* Wo       = (const float*)d_in[10];
  const float* bo       = (const float*)d_in[11];
  const float* gap_emb  = (const float*)d_in[12];
  const float* dur_emb  = (const float*)d_in[13];
  float* out = (float*)d_out;

  char* ws = (char*)d_ws;
  _Float16* xh  = (_Float16*)(ws);                         //  4 MiB: x in f16
  _Float16* WT  = (_Float16*)(ws + 4194304);               //  2 MiB: 4x W^T f16
  _Float16* Qd  = (_Float16*)(ws + 6291456);               //  4 MiB
  _Float16* Kd  = (_Float16*)(ws + 10485760);              //  4 MiB
  _Float16* Vt  = (_Float16*)(ws + 14680064);              //  4 MiB (transposed V)
  _Float16* ctx = (_Float16*)(ws + 18874368);              //  4 MiB

  k_cvt_w<<<4096, 256, 0, stream>>>(Wq, Wk, Wv, Wo, WT);
  k_cvt_x<<<8192, 256, 0, stream>>>(x, xh);
  k_proj<<<dim3(NTOK / 128, HIDN / 16, 3), 256, 0, stream>>>(xh, WT, bq, bk, bv, Qd, Kd, Vt);
  k_attn<<<dim3(BATCH * NHEAD, LSEQ / 128), 256, 0, stream>>>(Qd, Kd, Vt, amask, gids, dids,
                                                              gap_emb, dur_emb, ctx);
  k_oproj<<<dim3(NTOK / 128, HIDN / 16), 256, 0, stream>>>(ctx, WT + 3 * 512 * 512, bo, out);
}
